// ATNLPmodel_26010321944674
// MI455X (gfx1250) — matrix operand order repaired and checked by reading, not compiled, not run on hardware
//
#include <hip/hip_runtime.h>
#include <cstddef>

// ---------------------------------------------------------------------------
// Problem constants (from reference)
// ---------------------------------------------------------------------------
#define NB    8        // batch
#define NR    8        // snapshots per sequence
#define NC    256      // channels
#define LL1   512      // sequence length
#define LL2   32       // snapshot length
#define NQ    64       // NB*NR query rows
#define KDIM  8192     // NC*LL2 flattened feature dim
#define NDB   10000    // database rows
#define NTILE 16       // DB rows per block
#define KWAVE 2048     // KDIM / 4 waves

typedef __attribute__((ext_vector_type(16))) __bf16 v16bf;
typedef __attribute__((ext_vector_type(8)))  float  v8f;
typedef __attribute__((ext_vector_type(4)))  float  f32x4;   // clang-native, OK for NT builtins
typedef __attribute__((ext_vector_type(4)))  unsigned int u32x4;

// ---------------------------------------------------------------------------
// Kernel 1: gather-resample the 64 query windows, store bf16 Q (row-major
// [64][8192]) + per-row 1/(||q||+1e-8). Trivial cost (2 MB read).
// ---------------------------------------------------------------------------
__global__ __launch_bounds__(256) void build_q_kernel(
    const float* __restrict__ seq,   // (NB, NC, LL1)
    const int*   __restrict__ kp,    // (NB, NR, 2)
    __bf16*      __restrict__ Qbf,   // (NQ, KDIM) bf16
    float*       __restrict__ qinv)  // (NQ,)
{
  const int row = blockIdx.x;        // 0..63  == b*NR + r
  const int b   = row >> 3;
  const int tid = threadIdx.x;

  __shared__ int   spos[LL2];
  __shared__ float sred[256];

  if (tid < LL2) {
    int s = kp[row * 2 + 0];
    int e = kp[row * 2 + 1];
    e = (e > s + 1) ? e : (s + 1);
    spos[tid] = s + (tid * (e - s)) / LL2;   // floor div, matches reference
  }
  __syncthreads();

  const float* base = seq + (size_t)b * NC * LL1;
  float ss = 0.f;
  for (int idx = tid; idx < KDIM; idx += 256) {
    const int c = idx >> 5;          // channel
    const int l = idx & 31;          // position within snapshot
    const float v = base[c * LL1 + spos[l]];
    Qbf[(size_t)row * KDIM + idx] = (__bf16)v;
    ss = fmaf(v, v, ss);
  }

  sred[tid] = ss;
  __syncthreads();
  for (int off = 128; off > 0; off >>= 1) {
    if (tid < off) sred[tid] += sred[tid + off];
    __syncthreads();
  }
  if (tid == 0) qinv[row] = 1.f / (sqrtf(sred[0]) + 1e-8f);
}

// ---------------------------------------------------------------------------
// Kernel 2: fused cosine-sim GEMM. One non-temporal pass over the 327 MB
// database: raw dots via v_wmma_f32_16x16x32_bf16 + per-DB-row sum-of-
// squares, scaled at the end by qinv[m] * 1/(sqrt(sumsq)+1e-8).
//
// Block = 128 threads = 4 wave32 waves. Block owns 16 DB rows (N-tile);
// each wave owns a 2048-wide K chunk and all 64 query rows (4 M-tiles).
// DB stream is loaded NT so the reused bf16 Q stays hot in the 192 MB L2.
// Cross-wave reduction of the 64x16 partials + row sumsq via LDS atomics.
// ---------------------------------------------------------------------------
__global__ __launch_bounds__(128) void sim_gemm_kernel(
    const float*  __restrict__ db,    // (NDB, KDIM) f32
    const __bf16* __restrict__ Qbf,   // (NQ, KDIM) bf16
    const float*  __restrict__ qinv,  // (NQ,)
    float*        __restrict__ sim)   // (NQ, NDB)
{
  __shared__ float s_sim[NQ * NTILE];
  __shared__ float s_dsum[NTILE];

  const int tid  = threadIdx.x;
  const int wave = tid >> 5;
  const int lane = tid & 31;
  const int lm   = lane & 15;        // matrix row/col selector within fragment
  const int hi   = lane >> 4;        // half-wave selector
  const int n0   = blockIdx.x * NTILE;   // 10000 = 625*16, no tail

  for (int i = tid; i < NQ * NTILE; i += 128) s_sim[i] = 0.f;
  if (tid < NTILE) s_dsum[tid] = 0.f;
  __syncthreads();

  // B fragment (32x16 K-major, bf16): lane {lm,hi} holds DB row n0+lm,
  // K = kb + hi*16 + [0..15], packed 2 per dword (contiguous K order, same as A).
  const float* bptr = db + (size_t)(n0 + lm) * KDIM + wave * KWAVE + hi * 16;
  // A fragment (16x32, bf16): lane {lm,hi} holds query row m, two 16B chunks
  // at K = kb + hi*16 and K = kb + hi*16 + 8 (same K order as B).
  const int kA0 = wave * KWAVE + hi * 16;

  v8f   acc[4] = {};
  // 4 independent sumsq accumulators -> no serial v_fmac chain per K-step
  float ds0 = 0.f, ds1 = 0.f, ds2 = 0.f, ds3 = 0.f;

#pragma unroll 2
  for (int kk = 0; kk < KWAVE; kk += 32) {
    const f32x4* bp = (const f32x4*)(bptr + kk);
    // NT loads: database is touched exactly once -> don't thrash L2
    f32x4 f0 = __builtin_nontemporal_load(bp + 0);
    f32x4 f1 = __builtin_nontemporal_load(bp + 1);
    f32x4 f2 = __builtin_nontemporal_load(bp + 2);
    f32x4 f3 = __builtin_nontemporal_load(bp + 3);

    ds0 = fmaf(f0.x, f0.x, ds0); ds1 = fmaf(f0.y, f0.y, ds1);
    ds2 = fmaf(f0.z, f0.z, ds2); ds3 = fmaf(f0.w, f0.w, ds3);
    ds0 = fmaf(f1.x, f1.x, ds0); ds1 = fmaf(f1.y, f1.y, ds1);
    ds2 = fmaf(f1.z, f1.z, ds2); ds3 = fmaf(f1.w, f1.w, ds3);
    ds0 = fmaf(f2.x, f2.x, ds0); ds1 = fmaf(f2.y, f2.y, ds1);
    ds2 = fmaf(f2.z, f2.z, ds2); ds3 = fmaf(f2.w, f2.w, ds3);
    ds0 = fmaf(f3.x, f3.x, ds0); ds1 = fmaf(f3.y, f3.y, ds1);
    ds2 = fmaf(f3.z, f3.z, ds2); ds3 = fmaf(f3.w, f3.w, ds3);

    union { __bf16 h[16]; v16bf v; } bb;
    bb.h[0]  = (__bf16)f0.x; bb.h[1]  = (__bf16)f0.y;
    bb.h[2]  = (__bf16)f0.z; bb.h[3]  = (__bf16)f0.w;
    bb.h[4]  = (__bf16)f1.x; bb.h[5]  = (__bf16)f1.y;
    bb.h[6]  = (__bf16)f1.z; bb.h[7]  = (__bf16)f1.w;
    bb.h[8]  = (__bf16)f2.x; bb.h[9]  = (__bf16)f2.y;
    bb.h[10] = (__bf16)f2.z; bb.h[11] = (__bf16)f2.w;
    bb.h[12] = (__bf16)f3.x; bb.h[13] = (__bf16)f3.y;
    bb.h[14] = (__bf16)f3.z; bb.h[15] = (__bf16)f3.w;

#pragma unroll
    for (int mt = 0; mt < 4; ++mt) {
      const __bf16* ap = Qbf + (size_t)(mt * 16 + lm) * KDIM + kA0 + kk;
      union { u32x4 q[2]; v16bf v; } aa;
      aa.q[0] = *(const u32x4*)(ap);        // K = kb+hi*16 + 0..7
      aa.q[1] = *(const u32x4*)(ap + 8);    // K = kb+hi*16 + 8..15
      acc[mt] = __builtin_amdgcn_wmma_f32_16x16x32_bf16(
          false, aa.v, false, bb.v, (short)0, acc[mt], false, false);
    }
  }

  float dsum = (ds0 + ds1) + (ds2 + ds3);
  // lanes l and l^16 hold the two K-halves of the same DB row
  dsum += __shfl_xor(dsum, 16);
  if (hi == 0) atomicAdd(&s_dsum[lm], dsum);       // ds_add_f32

  // C/D layout: lane holds col n = lm, VGPR v holds row m = mt*16 + v + hi*8
#pragma unroll
  for (int mt = 0; mt < 4; ++mt) {
#pragma unroll
    for (int v = 0; v < 8; ++v) {
      atomicAdd(&s_sim[(mt * 16 + v + hi * 8) * NTILE + lm], acc[mt][v]);
    }
  }
  __syncthreads();

  for (int i = tid; i < NQ * NTILE; i += 128) {
    const int m = i >> 4;
    const int n = i & 15;
    const float dinv = 1.f / (sqrtf(s_dsum[n]) + 1e-8f);
    sim[(size_t)m * NDB + (n0 + n)] = s_sim[i] * qinv[m] * dinv;
  }
}

// ---------------------------------------------------------------------------
// Kernel 3: per-query-row max / argmax (first-index tie-break like jnp.argmax)
// ---------------------------------------------------------------------------
__global__ __launch_bounds__(256) void argmax_kernel(
    const float* __restrict__ sim,     // (NQ, NDB)
    const int*   __restrict__ cls,     // (NDB,)
    float*       __restrict__ unit,    // (NQ,)
    float*       __restrict__ tcls)    // (NQ,)
{
  __shared__ float sv[256];
  __shared__ int   si[256];
  const int m   = blockIdx.x;
  const int tid = threadIdx.x;

  const float* row = sim + (size_t)m * NDB;
  float best = -3.402823466e38f;
  int   bidx = 0;
  for (int i = tid; i < NDB; i += 256) {
    const float v = row[i];
    if (v > best) { best = v; bidx = i; }   // strided ascending -> first max
  }
  sv[tid] = best; si[tid] = bidx;
  __syncthreads();
  for (int off = 128; off > 0; off >>= 1) {
    if (tid < off) {
      const float v = sv[tid + off];
      const int   x = si[tid + off];
      if (v > sv[tid] || (v == sv[tid] && x < si[tid])) { sv[tid] = v; si[tid] = x; }
    }
    __syncthreads();
  }
  if (tid == 0) {
    unit[m] = sv[0];
    tcls[m] = (float)cls[si[0]];
  }
}

// ---------------------------------------------------------------------------
// Launcher
// ---------------------------------------------------------------------------
extern "C" void kernel_launch(void* const* d_in, const int* in_sizes, int n_in,
                              void* d_out, int out_size, void* d_ws, size_t ws_size,
                              hipStream_t stream) {
  (void)in_sizes; (void)n_in; (void)out_size; (void)ws_size;

  const float* seq = (const float*)d_in[0];   // (8,256,512) f32
  const int*   kp  = (const int*)  d_in[1];   // (8,8,2) int32
  const float* db  = (const float*)d_in[2];   // (10000,256,32) f32
  const int*   cls = (const int*)  d_in[3];   // (10000,) int32

  float* out  = (float*)d_out;
  float* sim  = out;                       // 64*10000
  float* unit = out + (size_t)NQ * NDB;    // 64
  float* tcls = unit + NQ;                 // 64 (class ids as float)

  // workspace: bf16 Q (1 MiB) + qinv (256 B)
  __bf16* Qbf  = (__bf16*)d_ws;
  float*  qinv = (float*)((char*)d_ws + (size_t)NQ * KDIM * sizeof(__bf16));

  build_q_kernel<<<NQ, 256, 0, stream>>>(seq, kp, Qbf, qinv);
  sim_gemm_kernel<<<NDB / NTILE, 128, 0, stream>>>(db, Qbf, qinv, sim);
  argmax_kernel<<<NQ, 256, 0, stream>>>(sim, cls, unit, tcls);
}
